// dinov2_decoder_2250562863364
// MI455X (gfx1250) — matrix operand order repaired and checked by reading, not compiled, not run on hardware
//
#include <hip/hip_runtime.h>
#include <stdint.h>

#define IN_CHN 1024
#define EMB_N  1024
#define OUT_N  768
#define BATCH  8
#define SEQ    2048

typedef __attribute__((ext_vector_type(16))) __bf16 v16bf;
typedef __attribute__((ext_vector_type(8)))  float  v8f;
typedef __attribute__((ext_vector_type(4)))  int    v4i;

union BFrag { v16bf v; uint4 q[2]; };
union Cvt4  { __bf16 h[4]; uint2 u; };

// ---- gfx1250 async global->LDS copy (ASYNCcnt-tracked), guarded -----------
#if defined(__has_builtin)
#  if __has_builtin(__builtin_amdgcn_global_load_async_to_lds_b128)
#    define HAVE_ASYNC_LDS 1
#  else
#    define HAVE_ASYNC_LDS 0
#  endif
#else
#  define HAVE_ASYNC_LDS 0
#endif

#if HAVE_ASYNC_LDS
typedef __attribute__((address_space(1))) v4i as1_v4i;
typedef __attribute__((address_space(3))) v4i as3_v4i;
#endif

__device__ __forceinline__ void async_copy16(const __bf16* g, __bf16* l) {
#if HAVE_ASYNC_LDS
    __builtin_amdgcn_global_load_async_to_lds_b128(
        (as1_v4i*)(uintptr_t)g,
        (as3_v4i*)(uint32_t)(uintptr_t)l,
        0, 0);
#else
    (void)g; (void)l;
#endif
}
__device__ __forceinline__ void wait_async() {
#if HAVE_ASYNC_LDS
#  if __has_builtin(__builtin_amdgcn_s_wait_asynccnt)
    __builtin_amdgcn_s_wait_asynccnt(0);
#  else
    asm volatile("s_wait_asynccnt 0" ::: "memory");
#  endif
#endif
}

__device__ __forceinline__ float wave_sum(float v) {
#pragma unroll
    for (int m = 16; m > 0; m >>= 1) v += __shfl_xor(v, m, 32);
    return v;
}
__device__ __forceinline__ float wave_max(float v) {
#pragma unroll
    for (int m = 16; m > 0; m >>= 1) v = fmaxf(v, __shfl_xor(v, m, 32));
    return v;
}

// ---------------------------------------------------------------------------
// LayerNorm (fp32 in -> bf16 out), one block per row, grid.y selects stream
// ---------------------------------------------------------------------------
__global__ __launch_bounds__(256) void ln_kernel(
    const float* __restrict__ img, const float* __restrict__ ref,
    const float* __restrict__ pose,
    const float* __restrict__ gamma, const float* __restrict__ beta,
    __bf16* __restrict__ ref_n, __bf16* __restrict__ pose_n,
    __bf16* __restrict__ img_n)
{
    const int row    = blockIdx.x;
    const int stream = blockIdx.y;
    const float* src = (stream == 0) ? ref : (stream == 1) ? pose : img;
    __bf16*      dst = (stream == 0) ? ref_n : (stream == 1) ? pose_n : img_n;
    src += (size_t)row * IN_CHN;
    dst += (size_t)row * IN_CHN;

    const int tid  = threadIdx.x;
    const int wave = tid >> 5;
    const int lane = tid & 31;

    float x[4];
    float s = 0.f, ss = 0.f;
#pragma unroll
    for (int i = 0; i < 4; ++i) {
        x[i] = src[tid + i * 256];
        s += x[i]; ss += x[i] * x[i];
    }
    s  = wave_sum(s);
    ss = wave_sum(ss);

    __shared__ float ws1[8], ws2[8];
    __shared__ float s_mu, s_rstd;
    if (lane == 0) { ws1[wave] = s; ws2[wave] = ss; }
    __syncthreads();
    if (tid == 0) {
        float ts = 0.f, tss = 0.f;
#pragma unroll
        for (int i = 0; i < 8; ++i) { ts += ws1[i]; tss += ws2[i]; }
        float mu  = ts * (1.0f / IN_CHN);
        float var = tss * (1.0f / IN_CHN) - mu * mu;
        s_mu = mu;
        s_rstd = rsqrtf(var + 1e-5f);
    }
    __syncthreads();
    const float mu = s_mu, rstd = s_rstd;
#pragma unroll
    for (int i = 0; i < 4; ++i) {
        const int c = tid + i * 256;
        dst[c] = (__bf16)((x[i] - mu) * rstd * gamma[c] + beta[c]);
    }
}

// ---------------------------------------------------------------------------
// Row softmax in place over bf16 scores, row length = SEQ (2048)
// ---------------------------------------------------------------------------
__global__ __launch_bounds__(256) void softmax_kernel(__bf16* __restrict__ scores)
{
    __bf16* row = scores + (size_t)blockIdx.x * SEQ;
    const int tid  = threadIdx.x;
    const int wave = tid >> 5;
    const int lane = tid & 31;

    float x[8];
    float m = -1e30f;
#pragma unroll
    for (int i = 0; i < 8; ++i) {
        x[i] = (float)row[tid + i * 256];
        m = fmaxf(m, x[i]);
    }
    m = wave_max(m);

    __shared__ float wred[8];
    __shared__ float s_m, s_sum;
    if (lane == 0) wred[wave] = m;
    __syncthreads();
    if (tid == 0) {
        float t = wred[0];
#pragma unroll
        for (int i = 1; i < 8; ++i) t = fmaxf(t, wred[i]);
        s_m = t;
    }
    __syncthreads();
    m = s_m;

    float s = 0.f;
#pragma unroll
    for (int i = 0; i < 8; ++i) { x[i] = __expf(x[i] - m); s += x[i]; }
    s = wave_sum(s);
    __syncthreads();              // wred reuse
    if (lane == 0) wred[wave] = s;
    __syncthreads();
    if (tid == 0) {
        float t = 0.f;
#pragma unroll
        for (int i = 0; i < 8; ++i) t += wred[i];
        s_sum = t;
    }
    __syncthreads();
    const float inv = 1.0f / s_sum;
#pragma unroll
    for (int i = 0; i < 8; ++i) row[tid + i * 256] = (__bf16)(x[i] * inv);
}

// ---------------------------------------------------------------------------
// Tiled WMMA GEMM: C[m,n] = alpha * (sum_k A[m,k]*B'[k,n] + bias[n]) (+resid)
//   B_NT   : B source is [N,K] row-major (ldb=K), else [K,N] row-major (ldb=N)
//   B_F32  : B source is fp32 (weights), converted to bf16 during LDS stage
//   OUT_F32: write fp32, else bf16
// Block: 256 threads (8 waves). Tile 128x128, K-step 32, double-buffered LDS.
// Wave grid 4(M) x 2(N); per wave 32x64 strip -> 8 WMMAs per K-step.
// bf16 tiles fetched with GLOBAL_LOAD_ASYNC_TO_LDS_B128 (ASYNCcnt overlap);
// f32-weight / transpose tiles staged through VGPRs into the inactive buffer.
// ---------------------------------------------------------------------------
template <bool B_NT, bool B_F32, bool OUT_F32, bool HAS_BIAS, bool HAS_RESID>
__global__ __launch_bounds__(256) void gemm_wmma(
    const __bf16* __restrict__ A, const void* __restrict__ Bsrc,
    const float* __restrict__ bias,
    const __bf16* __restrict__ r1, const __bf16* __restrict__ r2,
    void* __restrict__ Cout,
    int M, int N, int K,
    long long bsA, long long bsB, long long bsC, long long bsR,
    float alpha)
{
    constexpr bool USE_ASYNC   = HAVE_ASYNC_LDS != 0;
    constexpr bool B_ASYNC     = USE_ASYNC && B_NT && !B_F32;

    const int bz = blockIdx.z;
    A += (size_t)bz * bsA;
    const __bf16* Bb = (const __bf16*)Bsrc + (size_t)bz * bsB;
    const float*  Bf = (const float*)Bsrc  + (size_t)bz * bsB;
    float*  Cf = (float*)Cout  + (size_t)bz * bsC;
    __bf16* Ch = (__bf16*)Cout + (size_t)bz * bsC;
    if (HAS_RESID) { r1 += (size_t)bz * bsR; r2 += (size_t)bz * bsR; }

    const int tid     = threadIdx.x;
    const int lane    = tid & 31;
    const int wave    = tid >> 5;
    const int wave_m  = wave & 3;        // 4 M-strips of 32
    const int wave_n  = wave >> 2;       // 2 N-strips of 64
    const int lmod    = lane & 15;
    const int lhalf   = lane >> 4;
    const int rowBase = blockIdx.y * 128;
    const int colBase = blockIdx.x * 128;

    __shared__ __align__(16) __bf16 As[2][128][40];
    __shared__ __align__(16) __bf16 Bs[2][128][40];

    v8f acc[2][4];
#pragma unroll
    for (int mt = 0; mt < 2; ++mt)
#pragma unroll
        for (int nt = 0; nt < 4; ++nt)
            acc[mt][nt] = (v8f){0.f, 0.f, 0.f, 0.f, 0.f, 0.f, 0.f, 0.f};

    // ---- staging registers (fallback / convert / transpose paths) ----
    uint4  arq[2];
    uint4  brq[2];
    float4 brf[4];

    // Start fetching tile at k=kb into LDS buffer `buf`
    auto issueTiles = [&](int kb, int buf) {
        // A tile: 128 rows x 32 k, 16B per lane-chunk, 2 chunks/thread
#pragma unroll
        for (int it = 0; it < 2; ++it) {
            const int idx = tid + it * 256;          // 0..511
            const int r   = idx >> 2;
            const int c   = (idx & 3) * 8;
            const __bf16* g = A + (size_t)(rowBase + r) * K + kb + c;
            if constexpr (USE_ASYNC) async_copy16(g, &As[buf][r][c]);
            else                     arq[it] = *(const uint4*)g;
        }
        // B tile
        if constexpr (B_NT && B_F32) {
#pragma unroll
            for (int it = 0; it < 4; ++it) {
                const int idx = tid + it * 256;      // 0..1023
                const int n   = idx >> 3;
                const int c   = (idx & 7) * 4;
                brf[it] = *(const float4*)(Bf + (size_t)(colBase + n) * K + kb + c);
            }
        } else if constexpr (B_NT) {
#pragma unroll
            for (int it = 0; it < 2; ++it) {
                const int idx = tid + it * 256;
                const int n   = idx >> 2;
                const int c   = (idx & 3) * 8;
                const __bf16* g = Bb + (size_t)(colBase + n) * K + kb + c;
                if constexpr (B_ASYNC) async_copy16(g, &Bs[buf][n][c]);
                else                   brq[it] = *(const uint4*)g;
            }
        } else {
            // B is [K,N] row-major: 32 k-rows x 128 n (transpose on store)
#pragma unroll
            for (int it = 0; it < 2; ++it) {
                const int idx = tid + it * 256;      // 0..511
                const int k   = idx >> 4;
                const int n0  = (idx & 15) * 8;
                brq[it] = *(const uint4*)(Bb + (size_t)(kb + k) * N + colBase + n0);
            }
        }
    };
    // Commit staged registers into LDS buffer `buf`
    auto commitTiles = [&](int buf) {
        if constexpr (!USE_ASYNC) {
#pragma unroll
            for (int it = 0; it < 2; ++it) {
                const int idx = tid + it * 256;
                const int r   = idx >> 2;
                const int c   = (idx & 3) * 8;
                *(uint4*)(&As[buf][r][c]) = arq[it];
            }
        }
        if constexpr (B_NT && B_F32) {
#pragma unroll
            for (int it = 0; it < 4; ++it) {
                const int idx = tid + it * 256;
                const int n   = idx >> 3;
                const int c   = (idx & 7) * 4;
                Cvt4 cv;
                cv.h[0] = (__bf16)brf[it].x; cv.h[1] = (__bf16)brf[it].y;
                cv.h[2] = (__bf16)brf[it].z; cv.h[3] = (__bf16)brf[it].w;
                *(uint2*)(&Bs[buf][n][c]) = cv.u;
            }
        } else if constexpr (B_NT) {
            if constexpr (!B_ASYNC) {
#pragma unroll
                for (int it = 0; it < 2; ++it) {
                    const int idx = tid + it * 256;
                    const int n   = idx >> 2;
                    const int c   = (idx & 3) * 8;
                    *(uint4*)(&Bs[buf][n][c]) = brq[it];
                }
            }
        } else {
#pragma unroll
            for (int it = 0; it < 2; ++it) {
                const int idx = tid + it * 256;
                const int k   = idx >> 4;
                const int n0  = (idx & 15) * 8;
                const __bf16* h = (const __bf16*)&brq[it];
#pragma unroll
                for (int j = 0; j < 8; ++j) Bs[buf][n0 + j][k] = h[j];
            }
        }
    };
    auto compute = [&](int buf) {
        BFrag a[2];
        const int ak0 = lhalf * 8;                   // lanes 0-15: K0-7, 16-31: K8-15
#pragma unroll
        for (int mt = 0; mt < 2; ++mt) {
            const int am = wave_m * 32 + mt * 16 + lmod;
            a[mt].q[0] = *(const uint4*)(&As[buf][am][ak0]);
            a[mt].q[1] = *(const uint4*)(&As[buf][am][ak0 + 16]);
        }
        const int bk0 = lhalf * 8;                   // lanes 0-15: K0-7, 16-31: K8-15
#pragma unroll
        for (int nt = 0; nt < 4; ++nt) {
            BFrag b;
            const int bn = wave_n * 64 + nt * 16 + lmod;
            b.q[0] = *(const uint4*)(&Bs[buf][bn][bk0]);
            b.q[1] = *(const uint4*)(&Bs[buf][bn][bk0 + 16]);
#pragma unroll
            for (int mt = 0; mt < 2; ++mt)
                acc[mt][nt] = __builtin_amdgcn_wmma_f32_16x16x32_bf16(
                    false, a[mt].v, false, b.v, (short)0, acc[mt][nt], false, false);
        }
    };

    // ---- double-buffered, async-overlapped K loop ----
    const int nk = K >> 5;                           // K/32 steps
    issueTiles(0, 0);
    commitTiles(0);
    wait_async();
    __syncthreads();
    for (int i = 0; i < nk; ++i) {
        const int cur = i & 1;
        const int nxt = cur ^ 1;
        if (i + 1 < nk) issueTiles((i + 1) << 5, nxt);  // fetch next tile
        compute(cur);                                   // 8 WMMAs on current
        if (i + 1 < nk) {
            commitTiles(nxt);                           // staged paths only
            wait_async();                               // async tile landed
        }
        __syncthreads();
    }

    // ---- epilogue: C/D layout — VGPR r: M = r + 8*lhalf, N = lmod ----
#pragma unroll
    for (int mt = 0; mt < 2; ++mt) {
#pragma unroll
        for (int nt = 0; nt < 4; ++nt) {
            const int col = colBase + wave_n * 64 + nt * 16 + lmod;
            const float bval = HAS_BIAS ? bias[col] : 0.f;
#pragma unroll
            for (int r = 0; r < 8; ++r) {
                const int row = rowBase + wave_m * 32 + mt * 16 + lhalf * 8 + r;
                float v = (acc[mt][nt][r] + bval) * alpha;
                const size_t idx = (size_t)row * N + col;
                if (HAS_RESID) v += (float)r1[idx] + (float)r2[idx];
                if (OUT_F32) Cf[idx] = v;
                else         Ch[idx] = (__bf16)v;
            }
        }
    }
}

// ---------------------------------------------------------------------------
// Host launcher
// ---------------------------------------------------------------------------
extern "C" void kernel_launch(void* const* d_in, const int* in_sizes, int n_in,
                              void* d_out, int out_size, void* d_ws, size_t ws_size,
                              hipStream_t stream)
{
    const float* img   = (const float*)d_in[0];
    const float* ref   = (const float*)d_in[1];
    const float* pose  = (const float*)d_in[2];
    const float* gamma = (const float*)d_in[3];
    const float* beta  = (const float*)d_in[4];
    const float* Wq = (const float*)d_in[5];
    const float* bq = (const float*)d_in[6];
    const float* Wk = (const float*)d_in[7];
    const float* bk = (const float*)d_in[8];
    const float* Wv = (const float*)d_in[9];
    const float* bv = (const float*)d_in[10];
    const float* Wo = (const float*)d_in[11];
    const float* bo = (const float*)d_in[12];
    float* out = (float*)d_out;

    char* ws = (char*)d_ws;
    const size_t SZ = (size_t)BATCH * SEQ * EMB_N * sizeof(__bf16);   // 32 MB
    __bf16* ref_n  = (__bf16*)(ws + 0 * SZ);
    __bf16* pose_n = (__bf16*)(ws + 1 * SZ);
    __bf16* img_n  = (__bf16*)(ws + 2 * SZ);
    __bf16* Qb     = (__bf16*)(ws + 3 * SZ);
    __bf16* Kb     = (__bf16*)(ws + 4 * SZ);
    __bf16* Vb     = (__bf16*)(ws + 5 * SZ);
    __bf16* Ab     = (__bf16*)(ws + 6 * SZ);
    __bf16* Sc     = (__bf16*)(ws + 7 * SZ);                          // [B,S,S]

    const int Mtok = BATCH * SEQ;                                     // 16384
    const float scale = 0.03125f;                                     // EMB^-0.5

    // 1) LayerNorm all three streams
    ln_kernel<<<dim3(Mtok, 3), 256, 0, stream>>>(
        img, ref, pose, gamma, beta, ref_n, pose_n, img_n);

    // 2) Q/K/V projections (NT, fp32 weights). Q pre-scaled by EMB^-0.5.
    gemm_wmma<true, true, false, true, false>
        <<<dim3(EMB_N / 128, Mtok / 128, 1), 256, 0, stream>>>(
            ref_n, Wq, bq, nullptr, nullptr, Qb,
            Mtok, EMB_N, IN_CHN, 0, 0, 0, 0, scale);
    gemm_wmma<true, true, false, true, false>
        <<<dim3(EMB_N / 128, Mtok / 128, 1), 256, 0, stream>>>(
            pose_n, Wk, bk, nullptr, nullptr, Kb,
            Mtok, EMB_N, IN_CHN, 0, 0, 0, 0, 1.0f);
    gemm_wmma<true, true, false, true, false>
        <<<dim3(EMB_N / 128, Mtok / 128, 1), 256, 0, stream>>>(
            img_n, Wv, bv, nullptr, nullptr, Vb,
            Mtok, EMB_N, IN_CHN, 0, 0, 0, 0, 1.0f);

    // 3) scores = Q * K^T  (batched NT GEMM, bf16 x bf16, fully async-fed)
    gemm_wmma<true, false, false, false, false>
        <<<dim3(SEQ / 128, SEQ / 128, BATCH), 256, 0, stream>>>(
            Qb, Kb, nullptr, nullptr, nullptr, Sc,
            SEQ, SEQ, EMB_N,
            (long long)SEQ * EMB_N, (long long)SEQ * EMB_N,
            (long long)SEQ * SEQ, 0, 1.0f);

    // 4) softmax in place
    softmax_kernel<<<Mtok, 256, 0, stream>>>(Sc);

    // 5) attn = P * V + pose_n + img_n  (batched NN GEMM with residual)
    gemm_wmma<false, false, false, false, true>
        <<<dim3(EMB_N / 128, SEQ / 128, BATCH), 256, 0, stream>>>(
            Sc, Vb, nullptr, pose_n, img_n, Ab,
            SEQ, EMB_N, SEQ,
            (long long)SEQ * SEQ, (long long)SEQ * EMB_N,
            (long long)SEQ * EMB_N, (long long)SEQ * EMB_N, 1.0f);

    // 6) out = attn * Wo^T + bo  (NT, fp32 output)
    gemm_wmma<true, true, true, true, false>
        <<<dim3(OUT_N / 128, Mtok / 128, 1), 256, 0, stream>>>(
            Ab, Wo, bo, nullptr, nullptr, out,
            Mtok, OUT_N, EMB_N, 0, 0, 0, 0, 1.0f);
}
